// JMAC_MODEL_90005334655829
// MI455X (gfx1250) — matrix operand order repaired and checked by reading, not compiled, not run on hardware
//
#include <hip/hip_runtime.h>
#include <math.h>

#define NUM_ENT 50000
#define DIM 128
#define NUM_REL 1000
#define SLOPE 0.2f

typedef __attribute__((ext_vector_type(16))) _Float16 v16h;
typedef __attribute__((ext_vector_type(8)))  float    v8f;

// ---- CDNA5 WMMA fragment index helpers (wave32, v_wmma_f32_16x16x32_f16) ----
// A (16x32 f16): lanes 0-15 -> M=lane, K in {0..7,16..23}; lanes 16-31 -> M=lane-16, K in {8..15,24..31}
__device__ __forceinline__ int a_kidx(int lane, int e) {
    return ((e & 8) << 1) + ((lane >> 4) << 3) + (e & 7);
}
// B (32x16 f16): lanes 0-15 -> N=lane, K in {0..7,16..23}; lanes 16-31 -> N=lane-16, K in {8..15,24..31}
__device__ __forceinline__ int b_kidx(int lane, int e) {
    return ((e & 8) << 1) + ((lane >> 4) << 3) + (e & 7);
}
__device__ __forceinline__ float leaky_f(float x) { return x > 0.f ? x : SLOPE * x; }

union HU { unsigned short u; _Float16 h; };
__device__ __forceinline__ _Float16 lo16(unsigned int x) { HU u; u.u = (unsigned short)(x & 0xffffu); return u.h; }
__device__ __forceinline__ _Float16 hi16(unsigned int x) { HU u; u.u = (unsigned short)(x >> 16);     return u.h; }

// pack float4 -> 4 f16 and store as one 8-byte LDS write (dst 8B-aligned)
__device__ __forceinline__ void store4h(_Float16* dst, float4 w) {
    _Float16 h4[4] = { (_Float16)w.x, (_Float16)w.y, (_Float16)w.z, (_Float16)w.w };
    *(uint2*)dst = *(const uint2*)h4;
}

// A fragment from an LDS row of f16 (pairs e,e+1 are K-contiguous -> 32-bit LDS reads)
__device__ __forceinline__ v16h load_a_frag(const _Float16* row, int lane, int k0) {
    const unsigned int* r32 = (const unsigned int*)row;
    v16h a;
#pragma unroll
    for (int e = 0; e < 16; e += 2) {
        unsigned int pk = r32[(unsigned)(k0 + a_kidx(lane, e)) >> 1];
        a[e] = lo16(pk); a[e + 1] = hi16(pk);
    }
    return a;
}
// B fragment from transposed-weight LDS row wT[col][k] (pairs K-contiguous)
__device__ __forceinline__ v16h load_b_fragT(const _Float16* colrow, int lane, int k0) {
    const unsigned int* r32 = (const unsigned int*)colrow;
    v16h b;
#pragma unroll
    for (int e = 0; e < 16; e += 2) {
        unsigned int pk = r32[(unsigned)(k0 + b_kidx(lane, e)) >> 1];
        b[e] = lo16(pk); b[e + 1] = hi16(pk);
    }
    return b;
}
// stage one 128xK-chunk of W[K x 128] (f32, row-major) transposed into wT[128][128] f16
__device__ __forceinline__ void stage_wT(const float* __restrict__ W, int kc,
                                         _Float16 (*wT)[128], int t) {
    for (int idx = t; idx < 128 * 32; idx += 256) {
        int r = idx >> 5;                 // k-row within chunk
        int c4 = (idx & 31) << 2;         // col group of 4
        const float4 w4 = *(const float4*)&W[(unsigned)(kc + r) * 128 + c4];
        wT[c4 + 0][r] = (_Float16)w4.x;
        wT[c4 + 1][r] = (_Float16)w4.y;
        wT[c4 + 2][r] = (_Float16)w4.z;
        wT[c4 + 3][r] = (_Float16)w4.w;
    }
}

// =======================================================================
// Generic WMMA GEMM: out[M x 128] = act(concat_cols(A0|A1|A2)[M x K] @ W[K x 128])
// 256 threads (8 waves) per block; block owns 16 rows x 128 cols.
// A tile + transposed W chunk staged in LDS; K0/K1/K2 multiples of 128.
// =======================================================================
__global__ void __launch_bounds__(256)
gemm_wmma_k(const float* __restrict__ A0, int K0,
            const float* __restrict__ A1, int K1,
            const float* __restrict__ A2, int K2,
            const float* __restrict__ W, float* __restrict__ out,
            int M, int act) {
    __shared__ _Float16 aS[16][128];
    __shared__ _Float16 wT[128][128];
    int t = threadIdx.x;
    int lane = t & 31, wave = t >> 5;
    int mt = blockIdx.x;
    int K = K0 + K1 + K2;
    int cb = wave * 16;
    v8f c = {};
    for (int kc = 0; kc < K; kc += 128) {
        // chunk never straddles concat sources (all Ki multiples of 128): select once
        const float* src; int ld; int kb;
        if (kc < K0)            { src = A0; ld = K0; kb = kc; }
        else if (kc < K0 + K1)  { src = A1; ld = K1; kb = kc - K0; }
        else                    { src = A2; ld = K2; kb = kc - K0 - K1; }
        // stage A chunk: float4 loads, packed 8B LDS stores (512 groups / 256 thr)
        for (int idx = t; idx < 512; idx += 256) {
            int m = idx >> 5;
            int c4 = (idx & 31) << 2;
            int r = mt * 16 + m; if (r >= M) r = M - 1;
            float4 w4 = *(const float4*)&src[(unsigned)r * ld + kb + c4];
            store4h(&aS[m][c4], w4);
        }
        stage_wT(W, kc, wT, t);
        __syncthreads();
#pragma unroll
        for (int k0 = 0; k0 < 128; k0 += 32) {
            v16h a = load_a_frag(&aS[lane & 15][0], lane, k0);
            v16h b = load_b_fragT(&wT[cb + (lane & 15)][0], lane, k0);
            c = __builtin_amdgcn_wmma_f32_16x16x32_f16(false, a, false, b, (short)0, c, false, false);
        }
        __syncthreads();
    }
    if (mt * 16 + 16 <= M) {            // full tile: unguarded stores
#pragma unroll
        for (int v = 0; v < 8; ++v) {
            int r = mt * 16 + v + ((lane >> 4) << 3);
            float val = c[v];
            if (act) val = leaky_f(val);
            out[(unsigned)r * DIM + cb + (lane & 15)] = val;
        }
    } else {                             // ragged tail tile
#pragma unroll
        for (int v = 0; v < 8; ++v) {
            int r = mt * 16 + v + ((lane >> 4) << 3);
            if (r < M) {
                float val = c[v];
                if (act) val = leaky_f(val);
                out[(unsigned)r * DIM + cb + (lane & 15)] = val;
            }
        }
    }
}

// =======================================================================
// Edge attention score: block = 16 edges, 8 waves.
// comb = [x[i], x[j]*rel[t]] (f16, LDS); score = leaky(comb@watt)@aatt
// watt (256x128) staged transposed in two 128-K chunks; writes trans f16.
// =======================================================================
__global__ void __launch_bounds__(256)
edge_score_k(const int* __restrict__ rows, const int* __restrict__ cols,
             const int* __restrict__ ety,
             const float* __restrict__ x, const float* __restrict__ rel,
             const float* __restrict__ watt, const float* __restrict__ aatt,
             float* __restrict__ score, _Float16* __restrict__ transh, int E) {
    __shared__ int si[16], sj[16], st[16];
    __shared__ _Float16 comb[16][256];
    __shared__ _Float16 wT[128][128];
    __shared__ float sc[16];
    int t = threadIdx.x;
    int base = blockIdx.x * 16;
    if (t < 16) {
        int e = base + t; if (e >= E) e = E - 1;
        si[t] = rows[e]; sj[t] = cols[e]; st[t] = ety[e]; sc[t] = 0.f;
    }
    __syncthreads();
    // stage comb: float4 gathers; 16 rows x 64 float4-groups = 1024 groups
    for (int idx = t; idx < 1024; idx += 256) {
        int m = idx >> 6;
        int c4 = (idx & 63) << 2;
        if (c4 < 128) {
            float4 w4 = *(const float4*)&x[(unsigned)si[m] * DIM + c4];
            store4h(&comb[m][c4], w4);
        } else {
            int kk = c4 - 128;
            float4 a4 = *(const float4*)&x[(unsigned)sj[m] * DIM + kk];
            float4 r4 = *(const float4*)&rel[(unsigned)st[m] * DIM + kk];
            float4 w4 = make_float4(a4.x * r4.x, a4.y * r4.y, a4.z * r4.z, a4.w * r4.w);
            store4h(&comb[m][c4], w4);
            if (base + m < E) store4h(&transh[(unsigned)(base + m) * DIM + kk], w4);
        }
    }
    int lane = t & 31, wave = t >> 5;
    int cb = wave * 16;
    v8f c = {};
#pragma unroll
    for (int kc = 0; kc < 256; kc += 128) {
        __syncthreads();
        stage_wT(watt, kc, wT, t);
        __syncthreads();
#pragma unroll
        for (int k0 = 0; k0 < 128; k0 += 32) {
            v16h a = load_a_frag(&comb[lane & 15][0], lane, kc + k0);
            v16h b = load_b_fragT(&wT[cb + (lane & 15)][0], lane, k0);
            c = __builtin_amdgcn_wmma_f32_16x16x32_f16(false, a, false, b, (short)0, c, false, false);
        }
    }
    float av = aatt[cb + (lane & 15)];
#pragma unroll
    for (int v = 0; v < 8; ++v) {
        float h = leaky_f(c[v]);
        atomicAdd(&sc[v + ((lane >> 4) << 3)], h * av);
    }
    __syncthreads();
    if (t < 16 && base + t < E) score[base + t] = sc[t];
}

// =======================================================================
// Edge message: msg = (trans @ gcn) * coeff, scatter-add to accum[i]
// =======================================================================
__global__ void __launch_bounds__(256)
edge_msg_k(const int* __restrict__ rows, const _Float16* __restrict__ transh,
           const float* __restrict__ gcn, const float* __restrict__ coeff,
           float* __restrict__ accum, int E) {
    __shared__ int si[16];
    __shared__ float sco[16];
    __shared__ _Float16 at[16][128];
    __shared__ _Float16 wT[128][128];
    int t = threadIdx.x; int base = blockIdx.x * 16;
    if (t < 16) {
        int e = base + t; int ec = e >= E ? E - 1 : e;
        si[t] = rows[ec]; sco[t] = (e < E) ? coeff[ec] : 0.f;
    }
    // stage trans tile: 16x128 f16 = 256 x 16B -> one uint4 per thread
    if (base + 16 <= E) {
        const uint4* src = (const uint4*)(transh + (unsigned)base * DIM);
        uint4* dst = (uint4*)&at[0][0];
        dst[t] = src[t];
    } else {
        const unsigned int* src = (const unsigned int*)(transh + (unsigned)base * DIM);
        unsigned int* dst = (unsigned int*)&at[0][0];
        for (int idx = t; idx < 16 * 64; idx += 256) dst[idx] = src[idx];
    }
    stage_wT(gcn, 0, wT, t);
    __syncthreads();
    int lane = t & 31, wave = t >> 5; int cb = wave * 16;
    v8f c = {};
#pragma unroll
    for (int k0 = 0; k0 < 128; k0 += 32) {
        v16h a = load_a_frag(&at[lane & 15][0], lane, k0);
        v16h b = load_b_fragT(&wT[cb + (lane & 15)][0], lane, k0);
        c = __builtin_amdgcn_wmma_f32_16x16x32_f16(false, a, false, b, (short)0, c, false, false);
    }
#pragma unroll
    for (int v = 0; v < 8; ++v) {
        int m = v + ((lane >> 4) << 3);
        if (base + m < E)
            atomicAdd(&accum[(unsigned)si[m] * DIM + cb + (lane & 15)], c[v] * sco[m]);
    }
}

// =======================================================================
// Self-loop message: alpha ~= 1, no norm => accum[n] += (x[n]*rel_loop) @ gcn
// =======================================================================
__global__ void __launch_bounds__(256)
selfloop_k(const float* __restrict__ x, const float* __restrict__ relR,
           const float* __restrict__ gcn, float* __restrict__ accum, int M) {
    __shared__ _Float16 at[16][128];
    __shared__ _Float16 wT[128][128];
    int t = threadIdx.x; int base = blockIdx.x * 16;
    for (int idx = t; idx < 512; idx += 256) {
        int m = idx >> 5;
        int c4 = (idx & 31) << 2;
        int r = base + m; if (r >= M) r = M - 1;
        float4 a4 = *(const float4*)&x[(unsigned)r * DIM + c4];
        float4 r4 = *(const float4*)&relR[c4];
        store4h(&at[m][c4], make_float4(a4.x * r4.x, a4.y * r4.y, a4.z * r4.z, a4.w * r4.w));
    }
    stage_wT(gcn, 0, wT, t);
    __syncthreads();
    int lane = t & 31, wave = t >> 5; int cb = wave * 16;
    v8f c = {};
#pragma unroll
    for (int k0 = 0; k0 < 128; k0 += 32) {
        v16h a = load_a_frag(&at[lane & 15][0], lane, k0);
        v16h b = load_b_fragT(&wT[cb + (lane & 15)][0], lane, k0);
        c = __builtin_amdgcn_wmma_f32_16x16x32_f16(false, a, false, b, (short)0, c, false, false);
    }
    if (base + 16 <= M) {
#pragma unroll
        for (int v = 0; v < 8; ++v) {
            int r = base + v + ((lane >> 4) << 3);
            accum[(unsigned)r * DIM + cb + (lane & 15)] += c[v];
        }
    } else {
#pragma unroll
        for (int v = 0; v < 8; ++v) {
            int r = base + v + ((lane >> 4) << 3);
            if (r < M) accum[(unsigned)r * DIM + cb + (lane & 15)] += c[v];
        }
    }
}

// ---------------- small utility kernels ----------------
__device__ __forceinline__ void atomicMaxF(float* addr, float val) {
    unsigned int* ua = (unsigned int*)addr;
    unsigned int old = *ua;
    while (val > __uint_as_float(old)) {
        unsigned int prev = atomicCAS(ua, old, __float_as_uint(val));
        if (prev == old) break;
        old = prev;
    }
}
__global__ void fill_k(float* p, int n, float v) {
    int i = blockIdx.x * blockDim.x + threadIdx.x; if (i < n) p[i] = v;
}
__global__ void copy_k(const float* s, float* d, int n) {
    int i = blockIdx.x * blockDim.x + threadIdx.x; if (i < n) d[i] = s[i];
}
__global__ void deg_k(const int* rows, float* deg, int E) {
    int i = blockIdx.x * blockDim.x + threadIdx.x; if (i < E) atomicAdd(&deg[rows[i]], 1.f);
}
__global__ void dinv_k(float* deg, int N) {
    int i = blockIdx.x * blockDim.x + threadIdx.x;
    if (i < N) { float d = deg[i]; deg[i] = d > 0.f ? rsqrtf(d) : 0.f; }
}
__global__ void segmax_k(const int* rows, const float* score, float* m, int E) {
    int i = blockIdx.x * blockDim.x + threadIdx.x; if (i < E) atomicMaxF(&m[rows[i]], score[i]);
}
__global__ void fixm_k(float* m, int N) {
    int i = blockIdx.x * blockDim.x + threadIdx.x;
    if (i < N) { float v = m[i]; if (!(v >= -3.0e38f)) m[i] = 0.f; }
}
__global__ void expsum_k(const int* rows, const float* score, const float* m,
                         float* ev, float* s, int E) {
    int i = blockIdx.x * blockDim.x + threadIdx.x;
    if (i < E) { int r = rows[i]; float e = expf(score[i] - m[r]); ev[i] = e; atomicAdd(&s[r], e); }
}
__global__ void coeff_k(const int* rows, float* ev, const float* s, const float* dinv, int E) {
    int i = blockIdx.x * blockDim.x + threadIdx.x;
    if (i < E) { int r = rows[i]; ev[i] = ev[i] / (s[r] + 1e-16f) * dinv[r]; }
}
__global__ void l2norm_k(const float* __restrict__ x, float* __restrict__ y, int M) {
    int r = blockIdx.x; int t = threadIdx.x;           // blockDim = 128
    float v = x[(unsigned)r * DIM + t];
    __shared__ float red[128];
    red[t] = v * v; __syncthreads();
    for (int o = 64; o > 0; o >>= 1) { if (t < o) red[t] += red[t + o]; __syncthreads(); }
    float n = fmaxf(sqrtf(red[0]), 1e-12f);
    y[(unsigned)r * DIM + t] = v / n;
}
// batchnorm on (accum * 0.5) per column; population var; then tanh
__global__ void bn_stats_k(const float* __restrict__ a, float* meanv, float* istd, int M) {
    int c = blockIdx.x; int t = threadIdx.x;           // blockDim = 256
    float s = 0.f, ss = 0.f;
    for (int r = t; r < M; r += 256) { float v = a[(unsigned)r * DIM + c] * 0.5f; s += v; ss += v * v; }
    __shared__ float bs[256], bq[256];
    bs[t] = s; bq[t] = ss; __syncthreads();
    for (int o = 128; o > 0; o >>= 1) { if (t < o) { bs[t] += bs[t + o]; bq[t] += bq[t + o]; } __syncthreads(); }
    if (t == 0) {
        float m = bs[0] / M; meanv[c] = m;
        float var = bq[0] / M - m * m;
        istd[c] = rsqrtf(var + 1e-5f);
    }
}
__global__ void bn_apply_k(const float* __restrict__ a, const float* meanv,
                           const float* istd, float* __restrict__ out, int M) {
    int i = blockIdx.x * blockDim.x + threadIdx.x;
    if (i < M * DIM) {
        int c = i & (DIM - 1);
        out[i] = tanhf((a[i] * 0.5f - meanv[c]) * istd[c]);
    }
}

// =======================================================================
extern "C" void kernel_launch(void* const* d_in, const int* in_sizes, int n_in,
                              void* d_out, int out_size, void* d_ws, size_t ws_size,
                              hipStream_t stream) {
    const int E = in_sizes[1];
    const int N = NUM_ENT;
    const int* ei   = (const int*)d_in[0];
    const int* rows = ei;
    const int* cols = ei + E;
    const int* ety  = (const int*)d_in[1];
    const float* ent_info = (const float*)d_in[2];
    const float* ent_comp = (const float*)d_in[3];
    const float* rel_comp = (const float*)d_in[4];
    const float* rel_info = (const float*)d_in[5];
    const float* al11 = (const float*)d_in[6];
    const float* al12 = (const float*)d_in[7];
    const float* al21 = (const float*)d_in[8];
    const float* al22 = (const float*)d_in[9];
    const float* rl11 = (const float*)d_in[10];
    const float* rl12 = (const float*)d_in[11];
    const float* rl11a = (const float*)d_in[12];
    const float* rl12a = (const float*)d_in[13];
    const float* allc = (const float*)d_in[14];
    const float* cw1  = (const float*)d_in[15];
    const float* cw2  = (const float*)d_in[16];
    const float* cgcn = (const float*)d_in[17];
    const float* cloop = (const float*)d_in[18];
    const float* cwatt = (const float*)d_in[19];
    const float* caatt = (const float*)d_in[20];

    // ---- workspace carve (~260 MB) ----
    char* wb = (char*)d_ws; size_t off = 0;
    auto allocB = [&](size_t bytes) -> void* {
        void* p = wb + off; off = (off + bytes + 255) & ~(size_t)255; return p;
    };
    _Float16* transh = (_Float16*)allocB((size_t)E * DIM * 2);
    float* score  = (float*)allocB((size_t)E * 4);
    float* coef   = (float*)allocB((size_t)E * 4);
    float* segm   = (float*)allocB((size_t)N * 4);
    float* segs   = (float*)allocB((size_t)N * 4);
    float* dinv   = (float*)allocB((size_t)N * 4);
    float* accum  = (float*)allocB((size_t)N * DIM * 4);
    float* relcat = (float*)allocB((size_t)1008 * DIM * 4);
    float* reltmp = (float*)allocB((size_t)1008 * DIM * 4);
    float* relbuf = (float*)allocB((size_t)1008 * DIM * 4);
    float* tmpMK  = (float*)allocB((size_t)N * DIM * 4);
    float* initc  = (float*)allocB((size_t)N * DIM * 4);
    float* e1     = (float*)allocB((size_t)N * DIM * 4);
    float* ao1    = (float*)allocB((size_t)N * DIM * 4);
    float* e2     = (float*)allocB((size_t)N * DIM * 4);
    float* pos1n  = (float*)allocB((size_t)N * DIM * 4);
    float* ao2    = (float*)allocB((size_t)N * DIM * 4);
    float* ari    = (float*)allocB((size_t)1008 * DIM * 4);
    float* meanb  = (float*)allocB(DIM * 4);
    float* istdb  = (float*)allocB(DIM * 4);

    float* out = (float*)d_out;
    float* co1 = out + (size_t)N * DIM;           // second output
    float* pr1 = out + 2 * (size_t)N * DIM;       // third output

    auto cdiv = [](long a, long b) { return (unsigned)((a + b - 1) / b); };

    auto gemm = [&](const float* A0, int K0, const float* A1, int K1,
                    const float* A2, int K2, const float* W, float* O, int M, int act) {
        gemm_wmma_k<<<cdiv(M, 16), 256, 0, stream>>>(A0, K0, A1, K1, A2, K2, W, O, M, act);
    };

    // degree -> dinv (shared by all convs; depends only on edge_index)
    fill_k<<<cdiv(N, 256), 256, 0, stream>>>(dinv, N, 0.f);
    deg_k<<<cdiv(E, 256), 256, 0, stream>>>(rows, dinv, E);
    dinv_k<<<cdiv(N, 256), 256, 0, stream>>>(dinv, N);

    auto conv = [&](const float* x, const float* relin, int ci, float* op) {
        const float* w1 = cw1 + (size_t)ci * DIM * DIM;
        const float* w2 = cw2 + (size_t)ci * DIM * DIM;
        const float* gc = cgcn + (size_t)ci * DIM * DIM;
        const float* lp = cloop + (size_t)ci * DIM;
        const float* wa = cwatt + (size_t)ci * 2 * DIM * DIM;
        const float* aa = caatt + (size_t)ci * DIM;
        // rel = leaky(concat(rel_in, loop) @ w1) @ w2
        copy_k<<<cdiv(NUM_REL * DIM, 256), 256, 0, stream>>>(relin, relcat, NUM_REL * DIM);
        copy_k<<<1, 128, 0, stream>>>(lp, relcat + (size_t)NUM_REL * DIM, DIM);
        gemm(relcat, DIM, nullptr, 0, nullptr, 0, w1, reltmp, NUM_REL + 1, 1);
        gemm(reltmp, DIM, nullptr, 0, nullptr, 0, w2, relbuf, NUM_REL + 1, 0);
        // attention + scatter softmax
        fill_k<<<cdiv(N, 256), 256, 0, stream>>>(segm, N, -__builtin_inff());
        fill_k<<<cdiv(N, 256), 256, 0, stream>>>(segs, N, 0.f);
        fill_k<<<cdiv((long)N * DIM, 256), 256, 0, stream>>>(accum, N * DIM, 0.f);
        edge_score_k<<<cdiv(E, 16), 256, 0, stream>>>(rows, cols, ety, x, relbuf, wa, aa,
                                                      score, transh, E);
        segmax_k<<<cdiv(E, 256), 256, 0, stream>>>(rows, score, segm, E);
        fixm_k<<<cdiv(N, 256), 256, 0, stream>>>(segm, N);
        expsum_k<<<cdiv(E, 256), 256, 0, stream>>>(rows, score, segm, coef, segs, E);
        coeff_k<<<cdiv(E, 256), 256, 0, stream>>>(rows, coef, segs, dinv, E);
        // messages
        edge_msg_k<<<cdiv(E, 16), 256, 0, stream>>>(rows, transh, gc, coef, accum, E);
        selfloop_k<<<cdiv(N, 16), 256, 0, stream>>>(x, relbuf + (size_t)NUM_REL * DIM, gc, accum, N);
        // tanh(batchnorm((msg_n + msg_s)/2))
        bn_stats_k<<<DIM, 256, 0, stream>>>(accum, meanb, istdb, N);
        bn_apply_k<<<cdiv((long)N * DIM, 256), 256, 0, stream>>>(accum, meanb, istdb, op, N);
    };

    // forward_base
    l2norm_k<<<N, 128, 0, stream>>>(ent_comp, initc, N);
    gemm(initc, DIM, ent_info, DIM, nullptr, 0, al11, tmpMK, N, 1);
    gemm(tmpMK, DIM, nullptr, 0, nullptr, 0, al12, e1, N, 0);
    conv(e1, rel_info, 0, ao1);
    conv(ent_comp, rel_comp, 1, co1);
    l2norm_k<<<N, 128, 0, stream>>>(co1, pos1n, N);
    gemm(pos1n, DIM, ao1, DIM, nullptr, 0, al21, tmpMK, N, 1);
    gemm(tmpMK, DIM, nullptr, 0, nullptr, 0, al22, e2, N, 0);
    gemm(rel_comp, DIM, nullptr, 0, nullptr, 0, rl11, reltmp, NUM_REL, 1);
    gemm(reltmp, DIM, nullptr, 0, nullptr, 0, rl12, pr1, NUM_REL, 0);
    gemm(rel_info, DIM, nullptr, 0, nullptr, 0, rl11a, reltmp, NUM_REL, 1);
    gemm(reltmp, DIM, nullptr, 0, nullptr, 0, rl12a, ari, NUM_REL, 0);
    conv(e2, ari, 2, ao2);
    // out = concat(e1, ao1, ao2) @ all_linear_comp
    gemm(e1, DIM, ao1, DIM, ao2, DIM, allc, out, N, 0);
}
